// ERC_true_EFR_50749333569963
// MI455X (gfx1250) — matrix operand order repaired and checked by reading, not compiled, not run on hardware
//
#include <hip/hip_runtime.h>
#include <hip/hip_bf16.h>
#include <math.h>

// ---------------- problem constants ----------------
constexpr int CB = 64, CS = 256, CD = 1024, CH = 16, CF = 4096;
constexpr int CGH = 100, CNC = 7, CL = 4, CDH = 64;
constexpr int CM = CB * CS;           // 16384 rows of activations
constexpr int CW = 2 * CD + CGH;      // 2148 decoder input width

typedef __bf16 bf16_t;
typedef __attribute__((ext_vector_type(16))) __bf16 v16bf;
typedef __attribute__((ext_vector_type(8)))  __bf16 v8bf;
typedef __attribute__((ext_vector_type(8)))  float  v8f;
typedef unsigned int u32x4 __attribute__((ext_vector_type(4)));
typedef int          i32x8 __attribute__((ext_vector_type(8)));
typedef int          i32x4 __attribute__((ext_vector_type(4)));

__device__ __forceinline__ bf16_t f2bf(float f) {
  unsigned u = __builtin_bit_cast(unsigned, f);
  unsigned r = u + 0x7FFFu + ((u >> 16) & 1u);   // round-to-nearest-even
  unsigned short h = (unsigned short)(r >> 16);
  return __builtin_bit_cast(bf16_t, h);
}

// ---------------- fp32 -> bf16 conversion ----------------
__global__ void k_cvt_bf16(const float* __restrict__ in, bf16_t* __restrict__ out, int n) {
  int i = blockIdx.x * blockDim.x + threadIdx.x;
  int stride = gridDim.x * blockDim.x;
  for (; i < n; i += stride) out[i] = f2bf(in[i]);
}

// ---------------- embedding + sinusoidal PE ----------------
__global__ __launch_bounds__(256) void k_embed_pe(const int* __restrict__ src,
                                                  const float* __restrict__ emb,
                                                  float* __restrict__ xf,
                                                  bf16_t* __restrict__ xbf) {
  const int r = blockIdx.x;            // b*S + s
  const int s = r & (CS - 1);
  const int tok = src[r];
  const float scale = 32.0f;           // sqrt(1024)
  const float neg_ln1e4_over_D = -9.210340371976184f / (float)CD;
  int d0 = threadIdx.x * 4;
#pragma unroll
  for (int i = 0; i < 4; ++i) {
    int d = d0 + i;
    int half = d >> 1;
    float div = __expf((float)(2 * half) * neg_ln1e4_over_D);
    float ang = (float)s * div;
    float pe = (d & 1) ? __cosf(ang) : __sinf(ang);
    float v = emb[(size_t)tok * CD + d] * scale + pe;
    xf[(size_t)r * CD + d] = v;
    xbf[(size_t)r * CD + d] = f2bf(v);
  }
}

// ---------------- WMMA bf16 GEMM with TDM staging ----------------
// C[M,N] = A[M,K] * W[N,K]^T + bias ; tiles staged HBM -> LDS by the
// Tensor Data Mover (2D descriptor, pad feature gives the 96B LDS row
// stride: 64B tile row + 32B pad -> LDA = 48 bf16 elements).
constexpr int BM = 128, BN = 128, BK = 32;
constexpr int LDA = 48, LDB = 48;

// Issue one 2D TDM tile load: tile_dim = 32 x 128 bf16 from a row-major
// [rows, K] bf16 tensor into LDS at lds_addr (bytes, wave-relative).
__device__ __forceinline__ void tdm_tile_load(unsigned lds_addr,
                                              const bf16_t* gptr,
                                              int rows, int K) {
  unsigned long long ga = (unsigned long long)(uintptr_t)gptr;
  u32x4 g0;
  g0[0] = 1u;                                   // count=1, no gather
  g0[1] = lds_addr;                             // LDS byte address
  g0[2] = (unsigned)(ga & 0xFFFFFFFFu);         // global_addr[31:0]
  g0[3] = (unsigned)((ga >> 32) & 0x01FFFFFFu)  // global_addr[56:32]
          | (2u << 30);                         // type = 2 ("image")
  i32x8 g1;
  // data_size=1 (2B) | pad_enable | pad_interval=3 (16 DW) | pad_amount=7 (8 DW)
  g1[0] = (int)((1u << 16) | (1u << 20) | (3u << 22) | (7u << 25));
  g1[1] = (int)(((unsigned)K & 0xFFFFu) << 16);                 // tensor_dim0 lo
  g1[2] = (int)(((unsigned)K >> 16) |
                (((unsigned)rows & 0xFFFFu) << 16));            // dim0 hi | dim1 lo
  g1[3] = (int)(((unsigned)rows >> 16) | ((unsigned)BK << 16)); // dim1 hi | tile_dim0
  g1[4] = (int)(unsigned)BM;                                    // tile_dim1 | tile_dim2=0
  g1[5] = K;                                                    // tensor_dim0_stride lo
  g1[6] = 0;
  g1[7] = 0;
  i32x4 z4; z4[0] = 0; z4[1] = 0; z4[2] = 0; z4[3] = 0;
  i32x8 z8;
#pragma unroll
  for (int i = 0; i < 8; ++i) z8[i] = 0;
  __builtin_amdgcn_tensor_load_to_lds(g0, g1, z4, z4, z8, 0);
}

template <bool RELU, bool WF32, bool WBF>
__global__ __launch_bounds__(256)
void k_gemm_bf16(const bf16_t* __restrict__ A, const bf16_t* __restrict__ W,
                 const float* __restrict__ bias,
                 float* __restrict__ Cf, bf16_t* __restrict__ Cbf,
                 int M, int N, int K) {
  __shared__ __align__(32) bf16_t sA[2][BM * LDA];
  __shared__ __align__(32) bf16_t sB[2][BN * LDB];

  const int tid = threadIdx.x;
  const int w  = tid >> 5, l = tid & 31;
  const int wr = w >> 2, wc = w & 3;       // 2 x 4 wave grid
  const int lh = l >> 4, ll = l & 15;
  const int m0 = blockIdx.y * BM, n0 = blockIdx.x * BN;
  const bool issuer = (tid < 32);          // wave 0 drives the TDM

  v8f acc[4][2];
#pragma unroll
  for (int mi = 0; mi < 4; ++mi)
#pragma unroll
    for (int ni = 0; ni < 2; ++ni)
      acc[mi][ni] = (v8f){0.f, 0.f, 0.f, 0.f, 0.f, 0.f, 0.f, 0.f};

  if (issuer) {
    tdm_tile_load((unsigned)(uintptr_t)&sA[0][0], A + (size_t)m0 * K, M, K);
    tdm_tile_load((unsigned)(uintptr_t)&sB[0][0], W + (size_t)n0 * K, N, K);
  }

  int cur = 0;
  for (int kb = 0; kb < K; kb += BK) {
    if (issuer) {
      if (kb + BK < K) {
        // prefetch next tile pair into the other buffer, then wait for current
        tdm_tile_load((unsigned)(uintptr_t)&sA[cur ^ 1][0],
                      A + (size_t)m0 * K + (kb + BK), M, K);
        tdm_tile_load((unsigned)(uintptr_t)&sB[cur ^ 1][0],
                      W + (size_t)n0 * K + (kb + BK), N, K);
        __builtin_amdgcn_s_wait_tensorcnt(2);
      } else {
        __builtin_amdgcn_s_wait_tensorcnt(0);
      }
    }
    __syncthreads();

    // B fragments: lane holds col N = ll, K = lh*16 .. lh*16+15 (contiguous)
    v16bf bfrag[2];
#pragma unroll
    for (int ni = 0; ni < 2; ++ni) {
      int n = wc * 32 + ni * 16 + ll;
      bfrag[ni] = *(const v16bf*)&sB[cur][n * LDB + lh * 16];
    }
#pragma unroll
    for (int mi = 0; mi < 4; ++mi) {
      // A fragment: lane row M = ll, K = lh*16 .. lh*16+15 (contiguous)
      int r = wr * 64 + mi * 16 + ll;
      v8bf a0 = *(const v8bf*)&sA[cur][r * LDA + lh * 16];
      v8bf a1 = *(const v8bf*)&sA[cur][r * LDA + lh * 16 + 8];
      v16bf afrag = __builtin_shufflevector(a0, a1, 0, 1, 2, 3, 4, 5, 6, 7,
                                            8, 9, 10, 11, 12, 13, 14, 15);
#pragma unroll
      for (int ni = 0; ni < 2; ++ni) {
        acc[mi][ni] = __builtin_amdgcn_wmma_f32_16x16x32_bf16(
            false, afrag, false, bfrag[ni], (short)0, acc[mi][ni], false, false);
      }
    }
    __syncthreads();
    cur ^= 1;
  }

  // epilogue: C layout — lanes 0-15 rows j, lanes 16-31 rows j+8
#pragma unroll
  for (int ni = 0; ni < 2; ++ni) {
    int col = n0 + wc * 32 + ni * 16 + ll;
    float bv = bias[col];
#pragma unroll
    for (int mi = 0; mi < 4; ++mi) {
#pragma unroll
      for (int j = 0; j < 8; ++j) {
        int row = m0 + wr * 64 + mi * 16 + lh * 8 + j;
        float v = acc[mi][ni][j] + bv;
        if (RELU) v = fmaxf(v, 0.f);
        if (WF32) Cf[(size_t)row * N + col] = v;
        if (WBF)  Cbf[(size_t)row * N + col] = f2bf(v);
      }
    }
  }
}

// ---------------- causal attention (wave per query) ----------------
__global__ __launch_bounds__(256) void k_attention(const float* __restrict__ qkv,
                                                   bf16_t* __restrict__ obf) {
  __shared__ float qsh[8][CDH];
  __shared__ float probs[8][CS];
  const int w = threadIdx.x >> 5, l = threadIdx.x & 31;
  const int gw = blockIdx.x * 8 + w;
  const int qs = gw & (CS - 1);
  const int h  = (gw >> 8) & (CH - 1);
  const int b  = gw >> 12;
  const int row0 = b * CS;

  const float* qrow = qkv + (size_t)(row0 + qs) * (3 * CD) + h * CDH;
#pragma unroll
  for (int i = 0; i < 2; ++i) qsh[w][l + 32 * i] = qrow[l + 32 * i];
  __syncthreads();

  float sc[8];
  float mloc = -3.0e38f;
#pragma unroll
  for (int kt = 0; kt < 8; ++kt) {
    int ks = kt * 32 + l;
    float dot = -3.0e38f;
    if (ks <= qs) {
      const float* krow = qkv + (size_t)(row0 + ks) * (3 * CD) + CD + h * CDH;
      float a = 0.f;
      for (int d = 0; d < CDH; ++d) a += qsh[w][d] * krow[d];
      dot = a * 0.125f;                       // 1/sqrt(64)
    }
    sc[kt] = dot;
    mloc = fmaxf(mloc, dot);
  }
  for (int m = 16; m > 0; m >>= 1) mloc = fmaxf(mloc, __shfl_xor(mloc, m, 32));

  float ssum = 0.f;
#pragma unroll
  for (int kt = 0; kt < 8; ++kt) {
    int ks = kt * 32 + l;
    float p = (sc[kt] > -1.0e37f) ? __expf(sc[kt] - mloc) : 0.f;
    probs[w][ks] = p;
    ssum += p;
  }
  for (int m = 16; m > 0; m >>= 1) ssum += __shfl_xor(ssum, m, 32);
  float rinv = 1.f / ssum;
  __syncthreads();

  float a0 = 0.f, a1 = 0.f;
  const int d0 = 2 * l;
  for (int ks = 0; ks <= qs; ++ks) {
    float p = probs[w][ks];
    const float* vrow = qkv + (size_t)(row0 + ks) * (3 * CD) + 2 * CD + h * CDH;
    a0 += p * vrow[d0];
    a1 += p * vrow[d0 + 1];
  }
  a0 *= rinv; a1 *= rinv;
  bf16_t* orow = obf + (size_t)(row0 + qs) * CD + h * CDH;
  orow[d0]     = f2bf(a0);
  orow[d0 + 1] = f2bf(a1);
}

// ---------------- residual add + layernorm (block per row) ----------------
__global__ __launch_bounds__(256) void k_add_ln(const float* __restrict__ x,
                                                const float* __restrict__ y,
                                                const float* __restrict__ g,
                                                const float* __restrict__ bta,
                                                float* __restrict__ xo,
                                                bf16_t* __restrict__ xobf) {
  __shared__ float2 red[256];
  const int r = blockIdx.x, tid = threadIdx.x;
  const int d0 = tid * 4;
  float e[4];
  float s = 0.f, q = 0.f;
#pragma unroll
  for (int i = 0; i < 4; ++i) {
    float v = x[(size_t)r * CD + d0 + i] + y[(size_t)r * CD + d0 + i];
    e[i] = v; s += v; q += v * v;
  }
  red[tid] = make_float2(s, q);
  __syncthreads();
  for (int off = 128; off > 0; off >>= 1) {
    if (tid < off) {
      red[tid].x += red[tid + off].x;
      red[tid].y += red[tid + off].y;
    }
    __syncthreads();
  }
  float mean = red[0].x * (1.f / CD);
  float var  = red[0].y * (1.f / CD) - mean * mean;
  float rstd = rsqrtf(var + 1e-5f);
#pragma unroll
  for (int i = 0; i < 4; ++i) {
    int d = d0 + i;
    float v = (e[i] - mean) * rstd * g[d] + bta[d];
    xo[(size_t)r * CD + d] = v;
    xobf[(size_t)r * CD + d] = f2bf(v);
  }
}

// ---------------- GRU scan (block per batch element) ----------------
__global__ __launch_bounds__(320) void k_gru(const float* __restrict__ em,
                                             const float* __restrict__ Wih,
                                             const float* __restrict__ Whh,
                                             const float* __restrict__ bih,
                                             const float* __restrict__ bhh,
                                             float* __restrict__ emo) {
  __shared__ float h[CGH];
  __shared__ float hp[3 * CGH];
  __shared__ float xt[3 * CGH];
  const int b = blockIdx.x, t = threadIdx.x;
  if (t < CGH) h[t] = 0.f;
  __syncthreads();
  for (int s = 0; s < CS; ++s) {
    if (t < 3 * CGH) {
      float a = bhh[t];
      const float* wr_ = Whh + t * CGH;
      for (int j = 0; j < CGH; ++j) a += wr_[j] * h[j];
      hp[t] = a;
      float xa = bih[t];
      const float* e = em + (size_t)(b * CS + s) * 7;
      const float* wi = Wih + t * 7;
#pragma unroll
      for (int j = 0; j < 7; ++j) xa += wi[j] * e[j];
      xt[t] = xa;
    }
    __syncthreads();
    if (t < CGH) {
      float rg = 1.f / (1.f + __expf(-(xt[t] + hp[t])));
      float zg = 1.f / (1.f + __expf(-(xt[t + CGH] + hp[t + CGH])));
      float ng = tanhf(xt[t + 2 * CGH] + rg * hp[t + 2 * CGH]);
      float hn = (1.f - zg) * ng + zg * h[t];
      h[t] = hn;
      emo[(size_t)(b * CS + s) * CGH + t] = hn;
    }
    __syncthreads();
  }
}

// ---------------- main-token dot precompute (block per b, wave per class) --------
__global__ __launch_bounds__(256) void k_maindot(const float* __restrict__ x,
                                                 const float* __restrict__ decW,
                                                 const int* __restrict__ d_ids,
                                                 const int* __restrict__ ut_len,
                                                 float* __restrict__ maindot) {
  const int b = blockIdx.x;
  const int w = threadIdx.x >> 5, l = threadIdx.x & 31;
  const int len = ut_len[d_ids[b]];
  const float* xr = x + (size_t)(b * CS + (len - 1)) * CD;
  if (w < CNC) {
    const float* wr_ = decW + (size_t)w * CW + CD;
    float a = 0.f;
    for (int d = l; d < CD; d += 32) a += wr_[d] * xr[d];
    for (int m = 16; m > 0; m >>= 1) a += __shfl_xor(a, m, 32);
    if (l == 0) maindot[b * CNC + w] = a;
  }
}

// ---------------- decoder head (wave per token) ----------------
__global__ __launch_bounds__(256) void k_decoder(const float* __restrict__ x,
                                                 const float* __restrict__ emo,
                                                 const float* __restrict__ decW,
                                                 const float* __restrict__ decb,
                                                 const int* __restrict__ d_ids,
                                                 const int* __restrict__ ut_len,
                                                 const float* __restrict__ maindot,
                                                 float* __restrict__ out) {
  const int gw = blockIdx.x * 8 + (threadIdx.x >> 5);
  const int l = threadIdx.x & 31;
  const int b = gw >> 8, s = gw & (CS - 1);
  const int len = ut_len[d_ids[b]];
  const bool valid = s < len;
  const float* xr = x + (size_t)gw * CD;
  const float* er = emo + (size_t)gw * CGH;
  for (int c = 0; c < CNC; ++c) {
    const float* wr_ = decW + (size_t)c * CW;
    float a = 0.f;
    if (valid)
      for (int d = l; d < CD; d += 32) a += wr_[d] * xr[d];
    for (int gI = l; gI < CGH; gI += 32) a += wr_[2 * CD + gI] * er[gI];
    for (int m = 16; m > 0; m >>= 1) a += __shfl_xor(a, m, 32);
    if (l == 0)
      out[(size_t)gw * CNC + c] = a + decb[c] + (valid ? maindot[b * CNC + c] : 0.f);
  }
}

// ---------------- host orchestration ----------------
extern "C" void kernel_launch(void* const* d_in, const int* in_sizes, int n_in,
                              void* d_out, int out_size, void* d_ws, size_t ws_size,
                              hipStream_t stream) {
  (void)in_sizes; (void)n_in; (void)out_size; (void)ws_size;
  const int*   src    = (const int*)  d_in[0];
  const float* em_seq = (const float*)d_in[1];
  const int*   d_ids  = (const int*)  d_in[2];
  const int*   ut_len = (const int*)  d_in[3];
  const float* emb    = (const float*)d_in[4];
  const float* Wqkv   = (const float*)d_in[5];
  const float* bqkv   = (const float*)d_in[6];
  const float* Wo     = (const float*)d_in[7];
  const float* bo     = (const float*)d_in[8];
  const float* W1     = (const float*)d_in[9];
  const float* b1     = (const float*)d_in[10];
  const float* W2     = (const float*)d_in[11];
  const float* b2     = (const float*)d_in[12];
  const float* ln1_g  = (const float*)d_in[13];
  const float* ln1_b  = (const float*)d_in[14];
  const float* ln2_g  = (const float*)d_in[15];
  const float* ln2_b  = (const float*)d_in[16];
  const float* gWih   = (const float*)d_in[17];
  const float* gWhh   = (const float*)d_in[18];
  const float* gbih   = (const float*)d_in[19];
  const float* gbhh   = (const float*)d_in[20];
  const float* decW   = (const float*)d_in[21];
  const float* decb   = (const float*)d_in[22];
  float* out = (float*)d_out;

  char* ws = (char*)d_ws;
  size_t off = 0;
  auto alloc = [&](size_t bytes) -> void* {
    void* p = ws + off;
    off += (bytes + 255) & ~(size_t)255;
    return p;
  };
  const size_t nWqkv = (size_t)CL * 3 * CD * CD;
  const size_t nWo   = (size_t)CL * CD * CD;
  const size_t nW1   = (size_t)CL * CF * CD;
  const size_t nW2   = (size_t)CL * CD * CF;

  bf16_t* wqkv_bf = (bf16_t*)alloc(nWqkv * 2);
  bf16_t* wo_bf   = (bf16_t*)alloc(nWo * 2);
  bf16_t* w1_bf   = (bf16_t*)alloc(nW1 * 2);
  bf16_t* w2_bf   = (bf16_t*)alloc(nW2 * 2);
  float*  xf      = (float*) alloc((size_t)CM * CD * 4);
  bf16_t* xbf     = (bf16_t*)alloc((size_t)CM * CD * 2);
  float*  qkvbuf  = (float*) alloc((size_t)CM * 3 * CD * 4);
  bf16_t* h1bf    = (bf16_t*)alloc((size_t)CM * CF * 2);
  bf16_t* attnbf  = (bf16_t*)alloc((size_t)CM * CD * 2);
  float*  ybuf    = (float*) alloc((size_t)CM * CD * 4);
  float*  emo     = (float*) alloc((size_t)CM * CGH * 4);
  float*  maindot = (float*) alloc((size_t)CB * CNC * 4);

  // weight conversion fp32 -> bf16
  k_cvt_bf16<<<8192, 256, 0, stream>>>(Wqkv, wqkv_bf, (int)nWqkv);
  k_cvt_bf16<<<8192, 256, 0, stream>>>(Wo,   wo_bf,   (int)nWo);
  k_cvt_bf16<<<8192, 256, 0, stream>>>(W1,   w1_bf,   (int)nW1);
  k_cvt_bf16<<<8192, 256, 0, stream>>>(W2,   w2_bf,   (int)nW2);

  // embedding + positional encoding
  k_embed_pe<<<CM, 256, 0, stream>>>(src, emb, xf, xbf);

  for (int l = 0; l < CL; ++l) {
    // QKV projection: [16384,1024] x [3072,1024]^T -> fp32
    k_gemm_bf16<false, true, false><<<dim3(3 * CD / BN, CM / BM), 256, 0, stream>>>(
        xbf, wqkv_bf + (size_t)l * 3 * CD * CD, bqkv + (size_t)l * 3 * CD,
        qkvbuf, nullptr, CM, 3 * CD, CD);
    // causal attention -> bf16 context
    k_attention<<<CB * CH * CS / 8, 256, 0, stream>>>(qkvbuf, attnbf);
    // O projection
    k_gemm_bf16<false, true, false><<<dim3(CD / BN, CM / BM), 256, 0, stream>>>(
        attnbf, wo_bf + (size_t)l * CD * CD, bo + (size_t)l * CD,
        ybuf, nullptr, CM, CD, CD);
    // x = LN(x + attn_proj)
    k_add_ln<<<CM, 256, 0, stream>>>(xf, ybuf, ln1_g + (size_t)l * CD,
                                     ln1_b + (size_t)l * CD, xf, xbf);
    // FFN up + ReLU -> bf16 only
    k_gemm_bf16<true, false, true><<<dim3(CF / BN, CM / BM), 256, 0, stream>>>(
        xbf, w1_bf + (size_t)l * CF * CD, b1 + (size_t)l * CF,
        nullptr, h1bf, CM, CF, CD);
    // FFN down -> fp32
    k_gemm_bf16<false, true, false><<<dim3(CD / BN, CM / BM), 256, 0, stream>>>(
        h1bf, w2_bf + (size_t)l * CD * CF, b2 + (size_t)l * CD,
        ybuf, nullptr, CM, CD, CF);
    // x = LN(x + ffn)
    k_add_ln<<<CM, 256, 0, stream>>>(xf, ybuf, ln2_g + (size_t)l * CD,
                                     ln2_b + (size_t)l * CD, xf, xbf);
  }

  // GRU side branch and decoder head
  k_maindot<<<CB, 256, 0, stream>>>(xf, decW, d_ids, ut_len, maindot);
  k_gru<<<CB, 320, 0, stream>>>(em_seq, gWih, gWhh, gbih, gbhh, emo);
  k_decoder<<<CM / 8, 256, 0, stream>>>(xf, emo, decW, decb, d_ids, ut_len,
                                        maindot, out);
}
